// MHAttn_14594298872550
// MI455X (gfx1250) — matrix operand order repaired and checked by reading, not compiled, not run on hardware
//
#include <hip/hip_runtime.h>

// ---------------------------------------------------------------------------
// MHA with QLEN==1 on gfx1250 (wave32, WMMA bf16 16x16x32, f32 accumulate).
// ---------------------------------------------------------------------------

#define LDV 1032  // LDS row stride in ushorts: 2064B -> dword stride 516, 516%64=4 -> conflict-free

typedef __attribute__((ext_vector_type(16))) __bf16        v16bf;
typedef __attribute__((ext_vector_type(8)))  float          v8f;
typedef __attribute__((ext_vector_type(16))) unsigned short v16u;
typedef __attribute__((ext_vector_type(8)))  unsigned short v8u;

// Native f32 -> bf16 (hardware RNE conversion; avoids manual round sequences)
__device__ __forceinline__ unsigned short f2bf(float f) {
  __bf16 h = (__bf16)f;
  return __builtin_bit_cast(unsigned short, h);
}
__device__ __forceinline__ unsigned int pk2bf(float x, float y) {
  return (unsigned int)f2bf(x) | ((unsigned int)f2bf(y) << 16);
}
__device__ __forceinline__ v8f zero8() {
  v8f z = {0.f,0.f,0.f,0.f,0.f,0.f,0.f,0.f};
  return z;
}
__device__ __forceinline__ v8f wmma_bf16(v16u a, v16u b, v8f c) {
  return __builtin_amdgcn_wmma_f32_16x16x32_bf16(
      false, __builtin_bit_cast(v16bf, a),
      false, __builtin_bit_cast(v16bf, b),
      (short)0, c, false, false);
}

// A fragment (16x32 bf16) from LDS row-major [16][LDV].
// lane<16: row=lane,    K = {0..7, 16..23}   (hi8 = 0)
// lane>=16: row=lane-16, K = {8..15, 24..31} (hi8 = 8)
__device__ __forceinline__ v16u load_a_lds(const unsigned short* lds, int row,
                                           int k0, int hi8) {
  const unsigned short* p = lds + row * LDV + k0 + hi8;
  union { v16u w; v8u h[2]; } u;
  u.h[0] = *(const v8u*)(p);
  u.h[1] = *(const v8u*)(p + 16);
  return u.w;
}

// ---------------------------------------------------------------------------
// K1: convert Wv, Wo (f32 [1024][1024], row=K, col=N) into bf16 WMMA
//     B-fragment layout: frag[(nTile*32 + kChunk)*32 + lane][16].
//     lane = (n&15) | (((jj>>3)&1)<<4), idx = (jj&7) | ((jj>>4)<<3)   (jj = K within 32-chunk)
// ---------------------------------------------------------------------------
__global__ __launch_bounds__(256) void k1_convw(const float* __restrict__ Wv,
                                                const float* __restrict__ Wo,
                                                unsigned short* __restrict__ WvF,
                                                unsigned short* __restrict__ WoF) {
  int idx = blockIdx.x * 256 + threadIdx.x;          // 0 .. 2*1048576-1
  int m   = idx >> 20;
  int e   = idx & 1048575;
  int j   = e >> 10;                                 // K
  int n   = e & 1023;                                // N
  float val = m ? Wo[e] : Wv[e];
  int nT  = n >> 4;
  int c   = j >> 5;
  int jj  = j & 31;
  int lane = (n & 15) | (((jj >> 3) & 1) << 4);
  int id   = (jj & 7) | ((jj >> 4) << 3);
  unsigned short* dst = m ? WoF : WvF;
  dst[(size_t)((nT * 32 + c) * 32 + lane) * 16 + id] = f2bf(val);
}

// ---------------------------------------------------------------------------
// K2: qh[b][hd] = bq[hd] + sum_j q[b][j] * Wq[j][hd]   (16 x 1024, tiny)
// ---------------------------------------------------------------------------
__global__ __launch_bounds__(256) void k2_qproj(const float* __restrict__ q,
                                                const float* __restrict__ Wq,
                                                const float* __restrict__ bq,
                                                float* __restrict__ qh) {
  int idx = blockIdx.x * 256 + threadIdx.x;          // 0..16383
  int b   = idx >> 10;
  int hd  = idx & 1023;
  const float* qb = q + b * 1024;
  float acc = bq[hd];
  for (int j = 0; j < 1024; ++j)
    acc = fmaf(qb[j], Wq[j * 1024 + hd], acc);
  qh[idx] = acc;
}

// ---------------------------------------------------------------------------
// K3: fold Wk into the query: u[b][j][h] = sum_d Wk[j][h*64+d] * qh[b][h*64+d]
//     written directly in bf16 B-fragment layout (per b: 32 chunks x 32 x 16).
//     (bk term is constant over s -> cancels in softmax, dropped.)
// ---------------------------------------------------------------------------
__global__ __launch_bounds__(256) void k3_ufold(const float* __restrict__ Wk,
                                                const float* __restrict__ qh,
                                                unsigned short* __restrict__ uF) {
  int idx = blockIdx.x * 256 + threadIdx.x;          // 0..262143
  int b   = idx >> 14;
  int rem = idx & 16383;
  int j   = rem >> 4;
  int h   = rem & 15;
  const float* wrow = Wk + (size_t)j * 1024 + (h << 6);
  const float* qv   = qh + b * 1024 + (h << 6);
  float a = 0.f;
#pragma unroll 8
  for (int d = 0; d < 64; ++d) a = fmaf(wrow[d], qv[d], a);
  int c  = j >> 5;
  int jj = j & 31;
  int lane = h | (((jj >> 3) & 1) << 4);
  int id   = (jj & 7) | ((jj >> 4) << 3);
  uF[(size_t)b * 16384 + ((c * 32 + lane) * 16 + id)] = f2bf(a);
}

// ---------------------------------------------------------------------------
// K4: scores[b][s][h] = (k[b][s][:] . u[b][:][h]) * 0.125
//     One wave = one 16-row s-tile, N=16 heads, K-loop 32 chunks of 32.
//     k loaded f32 and converted to bf16 A-fragments on the fly.
// ---------------------------------------------------------------------------
__global__ __launch_bounds__(256) void k4_scores(const float* __restrict__ kmat,
                                                 const unsigned short* __restrict__ uF,
                                                 float* __restrict__ scores) {
  const int t    = threadIdx.x;
  const int wave = t >> 5, lane = t & 31;
  const int b    = blockIdx.x >> 5;
  const int sc   = blockIdx.x & 31;
  const int s0   = sc * 128 + wave * 16;
  const int srow = s0 + (lane & 15);
  const int hi8  = (lane >> 4) << 3;
  const float* krow = kmat + ((size_t)b * 4096 + srow) * 1024 + hi8;
  const unsigned short* ub = uF + (size_t)b * 16384 + lane * 16;

  v8f acc = zero8();
  for (int c = 0; c < 32; ++c) {
    const float* p = krow + c * 32;
    float4 f0 = *(const float4*)(p + 0);
    float4 f1 = *(const float4*)(p + 4);
    float4 f2 = *(const float4*)(p + 16);
    float4 f3 = *(const float4*)(p + 20);
    union { v16u w; unsigned int u32[8]; } a;
    a.u32[0] = pk2bf(f0.x, f0.y);
    a.u32[1] = pk2bf(f0.z, f0.w);
    a.u32[2] = pk2bf(f1.x, f1.y);
    a.u32[3] = pk2bf(f1.z, f1.w);
    a.u32[4] = pk2bf(f2.x, f2.y);
    a.u32[5] = pk2bf(f2.z, f2.w);
    a.u32[6] = pk2bf(f3.x, f3.y);
    a.u32[7] = pk2bf(f3.z, f3.w);
    v16u bf = *(const v16u*)(ub + c * 512);
    acc = wmma_bf16(a.w, bf, acc);
  }
  const int h     = lane & 15;
  const int mbase = (lane >> 4) << 3;
#pragma unroll
  for (int r = 0; r < 8; ++r) {
    int s = s0 + r + mbase;
    scores[((size_t)b * 4096 + s) * 16 + h] = acc[r] * 0.125f;
  }
}

// ---------------------------------------------------------------------------
// K5: softmax over S=4096 per (b,h); attn layout [b][h][s].
// ---------------------------------------------------------------------------
__global__ __launch_bounds__(256) void k5_softmax(const float* __restrict__ scores,
                                                  float* __restrict__ attn) {
  __shared__ float red[256];
  const int t = threadIdx.x;
  const int b = blockIdx.x >> 4;
  const int h = blockIdx.x & 15;
  float x[16];
  float mx = -3.0e38f;
#pragma unroll
  for (int j = 0; j < 16; ++j) {
    int s = t + j * 256;
    x[j] = scores[((size_t)b * 4096 + s) * 16 + h];
    mx = fmaxf(mx, x[j]);
  }
  red[t] = mx; __syncthreads();
  for (int o = 128; o > 0; o >>= 1) {
    if (t < o) red[t] = fmaxf(red[t], red[t + o]);
    __syncthreads();
  }
  mx = red[0]; __syncthreads();
  float sum = 0.f;
#pragma unroll
  for (int j = 0; j < 16; ++j) { x[j] = __expf(x[j] - mx); sum += x[j]; }
  red[t] = sum; __syncthreads();
  for (int o = 128; o > 0; o >>= 1) {
    if (t < o) red[t] += red[t + o];
    __syncthreads();
  }
  float inv = 1.f / red[0];
#pragma unroll
  for (int j = 0; j < 16; ++j)
    attn[((size_t)b * 16 + h) * 4096 + t + j * 256] = x[j] * inv;
}

// ---------------------------------------------------------------------------
// K6: fused V-projection -> per-head attn scale (+bv first) -> O-projection.
//     1 WG = 32 output rows (two 16-row A-tiles share every B-fragment ->
//     halves L2 weight traffic, 8 WMMA per 4 B-loads). 512 threads = 16
//     waves; wave w owns head w (columns [w*64, w*64+64) = 4 N-tiles).
// ---------------------------------------------------------------------------
__global__ __launch_bounds__(512) void k6_fused_vo(const float* __restrict__ vmat,
                                                   const unsigned short* __restrict__ WvF,
                                                   const unsigned short* __restrict__ WoF,
                                                   const float* __restrict__ bv,
                                                   const float* __restrict__ bo,
                                                   const float* __restrict__ attn,
                                                   float* __restrict__ out) {
  __shared__ unsigned short ldsV[32 * LDV];   // v tile, bf16       (66 KB)
  __shared__ unsigned short ldsH[32 * LDV];   // scaled vh tile     (66 KB)
  __shared__ float attnT[16 * 32];            // [h][m]             (2 KB)

  const int  t  = threadIdx.x;
  const long g0 = (long)blockIdx.x * 32;      // first global row (b*4096+s)

  // stage v tile (32 rows x 1024 f32) -> bf16 LDS, 64 elements per thread
  {
    int e0  = t * 64;
    int row = e0 >> 10, col = e0 & 1023;
    const float* src = vmat + (g0 + row) * 1024 + col;
    unsigned short* dst = ldsV + row * LDV + col;
#pragma unroll
    for (int q = 0; q < 64; q += 4) {
      float4 f = *(const float4*)(src + q);
      *(unsigned int*)(dst + q)     = pk2bf(f.x, f.y);
      *(unsigned int*)(dst + q + 2) = pk2bf(f.z, f.w);
    }
  }
  {  // attn for this tile's 32 rows x 16 heads: attnT[h*32+m]
    int h = t >> 5, m = t & 31;
    long g = g0 + m;
    attnT[h * 32 + m] = attn[(((g >> 12) << 4) + h) * 4096 + (g & 4095)];
  }
  __syncthreads();

  const int wave  = t >> 5, lane = t & 31;
  const int hi8   = (lane >> 4) << 3;   // A-frag K sub-offset
  const int arow  = lane & 15;          // A-frag row
  const int mbase = (lane >> 4) << 3;   // C/D row offset
  const int head  = wave;               // all 4 N-tiles of this wave = head `wave`

  // -------- GEMM1: VH = Vtile @ Wv --------
  v8f acc[2][4];
#pragma unroll
  for (int rt = 0; rt < 2; ++rt)
#pragma unroll
    for (int i = 0; i < 4; ++i) acc[rt][i] = zero8();

  for (int c = 0; c < 32; ++c) {
    v16u af0 = load_a_lds(ldsV,            arow, c * 32, hi8);  // rows  0-15
    v16u af1 = load_a_lds(ldsV + 16 * LDV, arow, c * 32, hi8);  // rows 16-31
#pragma unroll
    for (int i = 0; i < 4; ++i) {
      int nTile = wave * 4 + i;
      v16u bf = *(const v16u*)(WvF + (size_t)((nTile * 32 + c) * 32 + lane) * 16);
      acc[0][i] = wmma_bf16(af0, bf, acc[0][i]);
      acc[1][i] = wmma_bf16(af1, bf, acc[1][i]);
    }
  }
  // bias (before scale, per reference) + per-(row,head) attn scale -> ldsH bf16
#pragma unroll
  for (int i = 0; i < 4; ++i) {
    int ncol = (wave * 4 + i) * 16 + (lane & 15);
    float bvv = bv[ncol];
#pragma unroll
    for (int rt = 0; rt < 2; ++rt) {
#pragma unroll
      for (int r = 0; r < 8; ++r) {
        int row = rt * 16 + r + mbase;
        float x = (acc[rt][i][r] + bvv) * attnT[head * 32 + row];
        ldsH[row * LDV + ncol] = f2bf(x);
      }
    }
  }
  __syncthreads();

  // -------- GEMM2: OUT = scaled-VH @ Wo + bo --------
#pragma unroll
  for (int rt = 0; rt < 2; ++rt)
#pragma unroll
    for (int i = 0; i < 4; ++i) acc[rt][i] = zero8();

  for (int c = 0; c < 32; ++c) {
    v16u af0 = load_a_lds(ldsH,            arow, c * 32, hi8);
    v16u af1 = load_a_lds(ldsH + 16 * LDV, arow, c * 32, hi8);
#pragma unroll
    for (int i = 0; i < 4; ++i) {
      int nTile = wave * 4 + i;
      v16u bf = *(const v16u*)(WoF + (size_t)((nTile * 32 + c) * 32 + lane) * 16);
      acc[0][i] = wmma_bf16(af0, bf, acc[0][i]);
      acc[1][i] = wmma_bf16(af1, bf, acc[1][i]);
    }
  }
#pragma unroll
  for (int i = 0; i < 4; ++i) {
    int ncol = (wave * 4 + i) * 16 + (lane & 15);
    float bov = bo[ncol];
#pragma unroll
    for (int rt = 0; rt < 2; ++rt) {
#pragma unroll
      for (int r = 0; r < 8; ++r) {
        int row = rt * 16 + r + mbase;
        out[(g0 + row) * 1024 + ncol] = acc[rt][i][r] + bov;
      }
    }
  }
}

// ---------------------------------------------------------------------------
extern "C" void kernel_launch(void* const* d_in, const int* in_sizes, int n_in,
                              void* d_out, int out_size, void* d_ws, size_t ws_size,
                              hipStream_t stream) {
  const float* q  = (const float*)d_in[0];
  const float* k  = (const float*)d_in[1];
  const float* v  = (const float*)d_in[2];
  const float* Wq = (const float*)d_in[3];
  const float* bq = (const float*)d_in[4];
  const float* Wk = (const float*)d_in[5];
  // d_in[6] = bk: constant per (b,h) across keys -> cancels in softmax, unused.
  const float* Wv = (const float*)d_in[7];
  const float* bv = (const float*)d_in[8];
  const float* Wo = (const float*)d_in[9];
  const float* bo = (const float*)d_in[10];
  float* out = (float*)d_out;

  char* ws = (char*)d_ws;
  unsigned short* WvF    = (unsigned short*)(ws);                         // 2 MB
  unsigned short* WoF    = (unsigned short*)(ws + (size_t)(1u << 21));    // 2 MB
  float*          qh     = (float*)(ws + (size_t)(1u << 22));             // 64 KB
  unsigned short* uF     = (unsigned short*)(ws + (size_t)(1u << 22) + 65536);          // 512 KB
  float*          scores = (float*)(ws + (size_t)(1u << 22) + 65536 + 524288);          // 4 MB
  float*          attn   = (float*)(ws + (size_t)(1u << 22) + 65536 + 524288 + (size_t)(1u << 22)); // 4 MB

  (void)in_sizes; (void)n_in; (void)out_size; (void)ws_size;

  k1_convw   <<<8192, 256, 0, stream>>>(Wv, Wo, WvF, WoF);
  k2_qproj   <<<  64, 256, 0, stream>>>(q, Wq, bq, qh);
  k3_ufold   <<<1024, 256, 0, stream>>>(Wk, qh, uF);
  k4_scores  <<< 512, 256, 0, stream>>>(k, uF, scores);
  k5_softmax <<< 256, 256, 0, stream>>>(scores, attn);
  k6_fused_vo<<<2048, 512, 0, stream>>>(v, WvF, WoF, bv, bo, attn, out);
}
